// HeteroGNN_28140625724314
// MI455X (gfx1250) — matrix operand order repaired and checked by reading, not compiled, not run on hardware
//
#include <hip/hip_runtime.h>
#include <hip/hip_bf16.h>

// ---------------------------------------------------------------------------
// HeteroGNN forward for MI455X (gfx1250, wave32, WMMA).
// Scatter-aggregate via native float atomics (memory-bound, ~1.6GB/layer),
// dense transforms via fully-unrolled v_wmma_f32_16x16x32_f16 tiles with
// LDS-staged f16 weights (compile-time specialized per layer shape).
// ---------------------------------------------------------------------------

#define NNODES 131072

typedef __attribute__((ext_vector_type(16))) _Float16 v16h;
typedef __attribute__((ext_vector_type(8)))  float    v8f;

// ------------------------------- utility kernels ---------------------------

__global__ void zero_init_kernel(float* __restrict__ p, long n) {
  long i = (long)blockIdx.x * blockDim.x + threadIdx.x;
  if (i < n) p[i] = 0.f;
}

__global__ void neginf_init_kernel(float* __restrict__ p, long n) {
  long i = (long)blockIdx.x * blockDim.x + threadIdx.x;
  if (i < n) p[i] = -__builtin_inff();
}

__global__ void fixup_nonfinite_kernel(float* __restrict__ p, long n) {
  long i = (long)blockIdx.x * blockDim.x + threadIdx.x;
  if (i < n) {
    float v = p[i];
    if (!__builtin_isfinite(v)) p[i] = 0.f;
  }
}

// tile pe/router embeddings [64,16] -> [N,16]  (row r reads emb[r & 63])
__global__ void tile_emb_kernel(const float* __restrict__ emb,
                                float* __restrict__ out, int n) {
  int i = blockIdx.x * blockDim.x + threadIdx.x;
  if (i >= n) return;
  int row = i >> 4, f = i & 15;
  out[i] = emb[((row & 63) << 4) + f];
}

// ------------------------------- scatter kernels ---------------------------

__global__ void scatter_sum_kernel(const float* __restrict__ xsrc,
                                   const int* __restrict__ src,
                                   const int* __restrict__ dst,
                                   float* __restrict__ agg, int E, int din) {
  const int vpr = din >> 2;  // float4s per row
  long t = (long)blockIdx.x * blockDim.x + threadIdx.x;
  if (t >= (long)E * vpr) return;
  int e = (int)(t / vpr), v = (int)(t % vpr);
  int s = src[e], d = dst[e];
  float4 val = *(const float4*)(xsrc + (size_t)s * din + v * 4);
  float* o = agg + (size_t)d * din + v * 4;
  atomicAdd(o + 0, val.x);
  atomicAdd(o + 1, val.y);
  atomicAdd(o + 2, val.z);
  atomicAdd(o + 3, val.w);
}

__device__ inline void atomicMaxF(float* addr, float v) {
  // ordered-int trick; buffer initialized to -inf
  if (v >= 0.f) atomicMax((int*)addr, __float_as_int(v));
  else          atomicMin((unsigned int*)addr, __float_as_uint(v));
}

__global__ void scatter_max_kernel(const float* __restrict__ xsrc,
                                   const int* __restrict__ src,
                                   const int* __restrict__ dst,
                                   float* __restrict__ agg, int E, int din) {
  const int vpr = din >> 2;
  long t = (long)blockIdx.x * blockDim.x + threadIdx.x;
  if (t >= (long)E * vpr) return;
  int e = (int)(t / vpr), v = (int)(t % vpr);
  int s = src[e], d = dst[e];
  float4 val = *(const float4*)(xsrc + (size_t)s * din + v * 4);
  float* o = agg + (size_t)d * din + v * 4;
  atomicMaxF(o + 0, val.x);
  atomicMaxF(o + 1, val.y);
  atomicMaxF(o + 2, val.z);
  atomicMaxF(o + 3, val.w);
}

// ------------------------------- WMMA GEMM ---------------------------------
// A fragment: 16x32 f16, ISA layout:
//   lanes 0-15  : row M=lane,    halves 0..7 -> K=kk*32+0..7,  8..15 -> +16..23
//   lanes 16-31 : row M=lane-16, halves 0..7 -> K=kk*32+8..15, 8..15 -> +24..31
// Chunk0 (halves 0..7) starts at kk*32 + laneHi*8 : always < DIN.
// Chunk1 (halves 8..15) starts 16 higher: valid iff DIN >= 32.
template <int DIN>
__device__ inline v16h loadA(const float* __restrict__ x, int row0, int lane,
                             int kk) {
  const int laneHi = (lane >> 4) & 1;
  const int m = lane & 15;
  const float* p = x + (size_t)(row0 + m) * (size_t)DIN;
  v16h a;
  const int c0 = kk * 32 + laneHi * 8;
  {
    float4 f0 = *(const float4*)(p + c0);
    float4 f1 = *(const float4*)(p + c0 + 4);
    a[0] = (_Float16)f0.x; a[1] = (_Float16)f0.y;
    a[2] = (_Float16)f0.z; a[3] = (_Float16)f0.w;
    a[4] = (_Float16)f1.x; a[5] = (_Float16)f1.y;
    a[6] = (_Float16)f1.z; a[7] = (_Float16)f1.w;
  }
  if constexpr (DIN >= 32) {
    const int c1 = c0 + 16;
    float4 f0 = *(const float4*)(p + c1);
    float4 f1 = *(const float4*)(p + c1 + 4);
    a[8]  = (_Float16)f0.x; a[9]  = (_Float16)f0.y;
    a[10] = (_Float16)f0.z; a[11] = (_Float16)f0.w;
    a[12] = (_Float16)f1.x; a[13] = (_Float16)f1.y;
    a[14] = (_Float16)f1.z; a[15] = (_Float16)f1.w;
  } else {
#pragma unroll
    for (int h = 8; h < 16; ++h) a[h] = (_Float16)0.f;
  }
  return a;
}

// B fragment: 32x16 f16 from LDS weights stored transposed: sWt[col*64 + kidx]
//   lanes 0-15  : col N=lane,    halves 0..7 -> K=kk*32+0..7,  8..15 -> +16..23
//   lanes 16-31 : col N=lane-16, halves 0..7 -> K=kk*32+8..15, 8..15 -> +24..31
__device__ inline v16h loadB(const _Float16* __restrict__ sWt, int j, int lane,
                             int kk) {
  const int laneHi = (lane >> 4) & 1;
  const int col = j * 16 + (lane & 15);
  const _Float16* p = sWt + col * 64 + kk * 32 + laneHi * 8;
  v16h b;
#pragma unroll
  for (int h = 0; h < 16; ++h) b[h] = p[h < 8 ? h : h + 8];
  return b;
}

struct GemmArgs {
  const float* xdst;
  const float* agg0; const float* agg1; const float* agg2;
  const float* Wr0;  const float* Wr1;  const float* Wr2;
  const float* Wo0;  const float* Wo1;  const float* Wo2;
  const float* br0;  const float* br1;  const float* br2;
  float* out;
  int nRows;
};

template <int DIN, int DOUT, int K, bool CMAX, bool RELU>
__launch_bounds__(256)
__global__ void hetero_gemm_kernel(GemmArgs ga) {
  constexpr int KK = (DIN + 31) / 32;   // 1 (din=16) or 2 (din=64)
  constexpr int JT = (DOUT + 15) / 16;  // 4 (dout=64) or 1 (dout=2)
  constexpr float INVK = 1.0f / (float)K;

  // weights transposed to [col][kidx], zero-padded to 64x64, as f16 in LDS
  __shared__ _Float16 sW[2][K][64 * 64];  // [0]=Wr, [1]=Wo
  __shared__ float sbr[K][64];

  const int tid = threadIdx.x;
  const float* WrA[3] = {ga.Wr0, ga.Wr1, ga.Wr2};
  const float* WoA[3] = {ga.Wo0, ga.Wo1, ga.Wo2};
  const float* brA[3] = {ga.br0, ga.br1, ga.br2};
  const float* agA[3] = {ga.agg0, ga.agg1, ga.agg2};

#pragma unroll
  for (int k = 0; k < K; ++k) {
    for (int i = tid; i < 64 * 64; i += 256) {
      int col = i >> 6, kidx = i & 63;
      float vr = 0.f, vo = 0.f;
      if (kidx < DIN && col < DOUT) {
        vr = WrA[k][kidx * DOUT + col];
        vo = WoA[k][kidx * DOUT + col];
      }
      sW[0][k][col * 64 + kidx] = (_Float16)vr;
      sW[1][k][col * 64 + kidx] = (_Float16)vo;
    }
    for (int i = tid; i < 64; i += 256)
      sbr[k][i] = (i < DOUT) ? brA[k][i] : 0.f;
  }
  __syncthreads();

  const int lane   = tid & 31;
  const int wave   = tid >> 5;
  const int laneHi = (lane >> 4) & 1;
  const int lanelo = lane & 15;
  const int row0   = (blockIdx.x * 8 + wave) * 16;
  if (row0 >= ga.nRows) return;  // wave-uniform

  // root term A-fragments: shared by all K edge types, load once
  v16h aX[KK];
#pragma unroll
  for (int kk = 0; kk < KK; ++kk) aX[kk] = loadA<DIN>(ga.xdst, row0, lane, kk);

  float comb[JT][8];

#pragma unroll
  for (int k = 0; k < K; ++k) {
    v16h aA[KK];
#pragma unroll
    for (int kk = 0; kk < KK; ++kk)
      aA[kk] = loadA<DIN>(agA[k], row0, lane, kk);

#pragma unroll
    for (int j = 0; j < JT; ++j) {
      v8f c = {};
#pragma unroll
      for (int kk = 0; kk < KK; ++kk) {
        v16h b = loadB(&sW[0][k][0], j, lane, kk);
        c = __builtin_amdgcn_wmma_f32_16x16x32_f16(false, aA[kk], false, b,
                                                   (short)0, c, false, false);
      }
#pragma unroll
      for (int kk = 0; kk < KK; ++kk) {
        v16h b = loadB(&sW[1][k][0], j, lane, kk);
        c = __builtin_amdgcn_wmma_f32_16x16x32_f16(false, aX[kk], false, b,
                                                   (short)0, c, false, false);
      }
      const float bias = sbr[k][j * 16 + lanelo];
      if (k == 0) {
#pragma unroll
        for (int r = 0; r < 8; ++r) comb[j][r] = c[r] + bias;
      } else if (CMAX) {
#pragma unroll
        for (int r = 0; r < 8; ++r) comb[j][r] = fmaxf(comb[j][r], c[r] + bias);
      } else {
#pragma unroll
        for (int r = 0; r < 8; ++r) comb[j][r] += c[r] + bias;
      }
    }
  }

#pragma unroll
  for (int j = 0; j < JT; ++j) {
    const int col = j * 16 + lanelo;
#pragma unroll
    for (int r = 0; r < 8; ++r) {
      float v = comb[j][r];
      if (!CMAX) v *= INVK;
      if (RELU) v = fmaxf(v, 0.f);
      if (col < DOUT) {
        int row = row0 + laneHi * 8 + r;  // C/D: lanes0-15 M=r, 16-31 M=8+r
        ga.out[(size_t)row * DOUT + col] = v;
      }
    }
  }
}

template <int DIN, int DOUT, int K, bool CMAX, bool RELU>
static void launch_gemm(const GemmArgs& ga, int blocks, hipStream_t s) {
  hetero_gemm_kernel<DIN, DOUT, K, CMAX, RELU><<<blocks, 256, 0, s>>>(ga);
}

// ------------------------------- host orchestration ------------------------

extern "C" void kernel_launch(void* const* d_in, const int* in_sizes, int n_in,
                              void* d_out, int out_size, void* d_ws,
                              size_t ws_size, hipStream_t stream) {
  const int N = NNODES;

  const float* x_task_in  = (const float*)d_in[0];
  const float* pe_emb     = (const float*)d_in[8];
  const float* router_emb = (const float*)d_in[9];
  const float* params     = (const float*)d_in[10];

  // edges: d_in[1..7], each [2,E] int32 (row0=src, row1=dst)
  const int* eSrcP[7]; const int* eDstP[7]; int eCnt[7];
  for (int i = 0; i < 7; ++i) {
    int E = in_sizes[1 + i] / 2;
    const int* p = (const int*)d_in[1 + i];
    eSrcP[i] = p; eDstP[i] = p + E; eCnt[i] = E;
  }

  // ETYPES order: generates_for, requires, mapped_to, rev_mapped_to, link,
  //               interface, rev_interface
  const int eSrcT[7] = {0, 0, 0, 1, 2, 2, 1};  // 0=task 1=pe 2=router
  const int eIsMax[7] = {1, 0, 0, 0, 0, 0, 0};
  // alphabetical pytree key order within a layer dict
  const int sortedIdx[7] = {0, 4, 3, 6, 2, 1, 5};
  // incoming etypes per destination type (reference append order)
  const int groups[3][3] = {{0, 1, 3}, {2, 5, -1}, {4, 6, -1}};
  const int groupK[3] = {3, 2, 2};

  // workspace layout
  size_t off = 0;
  auto alloc = [&](size_t bytes) -> float* {
    float* p = (float*)((char*)d_ws + off);
    off += (bytes + 255) & ~(size_t)255;
    return p;
  };
  float* xbuf[2][3];
  for (int a = 0; a < 2; ++a)
    for (int t = 0; t < 3; ++t) xbuf[a][t] = alloc((size_t)N * 64 * sizeof(float));
  float* aggb[3];
  for (int k = 0; k < 3; ++k) aggb[k] = alloc((size_t)N * 64 * sizeof(float));

  // layer dims and param base offsets (pytree: per layer, sorted etype,
  // inner order Wo, Wr, br)
  const int dins[5]  = {16, 64, 64, 64, 64};
  const int douts[5] = {64, 64, 64, 64, 2};
  size_t layerBase[5]; size_t acc = 0;
  for (int li = 0; li < 5; ++li) {
    layerBase[li] = acc;
    acc += (size_t)7 * (2 * (size_t)dins[li] * douts[li] + douts[li]);
  }

  // initial pe/router features: tile [64,16] -> [N,16]
  {
    int n = N * 16;
    tile_emb_kernel<<<(n + 255) / 256, 256, 0, stream>>>(pe_emb, xbuf[0][1], n);
    tile_emb_kernel<<<(n + 255) / 256, 256, 0, stream>>>(router_emb, xbuf[0][2], n);
  }

  const float* cur[3] = {x_task_in, xbuf[0][1], xbuf[0][2]};

  for (int li = 0; li < 5; ++li) {
    const int din = dins[li], dout = douts[li];
    const int combineMax = (li == 1 || li == 3);

    float* nxt[3];
    if (li == 4) {
      nxt[0] = (float*)d_out;                 // task [N,2]
      nxt[1] = (float*)d_out + (size_t)N * 2; // pe   [N,2]
      nxt[2] = (float*)d_out + (size_t)N * 4; // router [N,2]
    } else {
      for (int t = 0; t < 3; ++t) nxt[t] = xbuf[(li + 1) & 1][t];
    }

    for (int t = 0; t < 3; ++t) {
      const int K = groupK[t];
      // aggregate each incoming edge type
      for (int k = 0; k < K; ++k) {
        const int e = groups[t][k];
        const long nAgg = (long)N * din;
        const int initBlocks = (int)((nAgg + 255) / 256);
        if (eIsMax[e])
          neginf_init_kernel<<<initBlocks, 256, 0, stream>>>(aggb[k], nAgg);
        else
          zero_init_kernel<<<initBlocks, 256, 0, stream>>>(aggb[k], nAgg);

        const long total = (long)eCnt[e] * (din >> 2);
        const int sBlocks = (int)((total + 255) / 256);
        if (eIsMax[e]) {
          scatter_max_kernel<<<sBlocks, 256, 0, stream>>>(
              cur[eSrcT[e]], eSrcP[e], eDstP[e], aggb[k], eCnt[e], din);
          fixup_nonfinite_kernel<<<initBlocks, 256, 0, stream>>>(aggb[k], nAgg);
        } else {
          scatter_sum_kernel<<<sBlocks, 256, 0, stream>>>(
              cur[eSrcT[e]], eSrcP[e], eDstP[e], aggb[k], eCnt[e], din);
        }
      }

      // fused WMMA GEMM + group combine + bias + relu
      GemmArgs ga{};
      ga.xdst = cur[t];
      ga.agg0 = aggb[0];
      ga.agg1 = (K > 1) ? aggb[1] : aggb[0];
      ga.agg2 = (K > 2) ? aggb[2] : aggb[0];
      const float* Wrp[3]; const float* Wop[3]; const float* brp[3];
      for (int k = 0; k < 3; ++k) {
        const int e = groups[t][k < K ? k : 0];
        const size_t slot = (size_t)sortedIdx[e] *
                            (2 * (size_t)din * dout + dout);
        const float* base = params + layerBase[li] + slot;
        Wop[k] = base;
        Wrp[k] = base + (size_t)din * dout;
        brp[k] = base + 2 * (size_t)din * dout;
      }
      ga.Wr0 = Wrp[0]; ga.Wr1 = Wrp[1]; ga.Wr2 = Wrp[2];
      ga.Wo0 = Wop[0]; ga.Wo1 = Wop[1]; ga.Wo2 = Wop[2];
      ga.br0 = brp[0]; ga.br1 = brp[1]; ga.br2 = brp[2];
      ga.out = nxt[t];
      ga.nRows = N;

      const int blocks = N / (8 * 16);  // 8 waves/block, 16 rows/wave
      if (li == 0) {
        if (K == 3) launch_gemm<16, 64, 3, false, true>(ga, blocks, stream);
        else        launch_gemm<16, 64, 2, false, true>(ga, blocks, stream);
      } else if (li == 4) {
        if (K == 3) launch_gemm<64, 2, 3, false, false>(ga, blocks, stream);
        else        launch_gemm<64, 2, 2, false, false>(ga, blocks, stream);
      } else if (combineMax) {
        if (K == 3) launch_gemm<64, 64, 3, true, true>(ga, blocks, stream);
        else        launch_gemm<64, 64, 2, true, true>(ga, blocks, stream);
      } else {
        if (K == 3) launch_gemm<64, 64, 3, false, true>(ga, blocks, stream);
        else        launch_gemm<64, 64, 2, false, true>(ga, blocks, stream);
      }
    }

    for (int t = 0; t < 3; ++t) cur[t] = nxt[t];
  }
}
